// MeshUpConv_37623913513299
// MI455X (gfx1250) — matrix operand order repaired and checked by reading, not compiled, not run on hardware
//
#include <hip/hip_runtime.h>
#include <hip/hip_bf16.h>
#include <stdint.h>

#define NN 50000
#define EE 400000
#define CIN 64
#define COUT 32
#define KK 9
#define YW 320   // 9*32 message cols + 32 root cols

typedef __attribute__((ext_vector_type(16))) __bf16 v16bf;
typedef __attribute__((ext_vector_type(8)))  float  v8f;

static __device__ __forceinline__ uint16_t f32_to_bf16(float f) {
    uint32_t u = __float_as_uint(f);
    uint32_t r = u + 0x7FFFu + ((u >> 16) & 1u);   // round-to-nearest-even
    return (uint16_t)(r >> 16);
}

// ---------------- basis: B[e,s] = q1[s/3]*q0[s%3], open quadratic B-spline ----
__global__ __launch_bounds__(256) void basis_kernel(const float* __restrict__ pseudo,
                                                    float* __restrict__ Bb) {
    int e = blockIdx.x * blockDim.x + threadIdx.x;
    if (e >= EE) return;
    float t0 = pseudo[2 * e], t1 = pseudo[2 * e + 1];
    float q0[3] = {0.5f * (1.f - t0) * (1.f - t0), -t0 * t0 + t0 + 0.5f, 0.5f * t0 * t0};
    float q1[3] = {0.5f * (1.f - t1) * (1.f - t1), -t1 * t1 + t1 + 0.5f, 0.5f * t1 * t1};
    float* o = Bb + (size_t)e * KK;
#pragma unroll
    for (int s = 0; s < KK; ++s) o[s] = q1[s / 3] * q0[s % 3];
}

// ---------------- fp32 -> bf16 bulk convert --------------------------------
__global__ __launch_bounds__(256) void cvt_bf16_kernel(const float* __restrict__ in,
                                                       uint16_t* __restrict__ out, int n) {
    int i = blockIdx.x * blockDim.x + threadIdx.x;
    if (i < n) out[i] = f32_to_bf16(in[i]);
}

// ---------------- build transposed bf16 weight  Wt[320][Cin] ----------------
// col j<288: Wt[j][c] = W[j/32][c][j%32];  col j>=288: root[c][j-288]
__global__ __launch_bounds__(256) void build_wt_kernel(const float* __restrict__ W,
                                                       const float* __restrict__ root,
                                                       uint16_t* __restrict__ Wt, int Cin) {
    int i = blockIdx.x * blockDim.x + threadIdx.x;
    if (i >= YW * Cin) return;
    int col = i / Cin, c = i % Cin;
    float v = (col < 288) ? W[(size_t)(col >> 5) * Cin * 32 + (size_t)c * 32 + (col & 31)]
                          : root[(size_t)c * 32 + (col - 288)];
    Wt[(size_t)col * Cin + c] = f32_to_bf16(v);
}

// ---------------- WMMA GEMM:  Y[M,320] = Abf[M,K] * Wt[320,K]^T  ------------
// K statically 32 or 64. One wave owns one 16-row M-tile and sweeps 4
// consecutive 16-col N-tiles, reusing the A fragment registers 4x.
template <int K>
__global__ __launch_bounds__(256) void gemm_wmma_kernel(const uint16_t* __restrict__ Abf,
                                                        const uint16_t* __restrict__ Wt,
                                                        float* __restrict__ Y,
                                                        int Mtiles) {
    constexpr int KT = K / 32;                 // WMMA steps along K (1 or 2)
    constexpr int NGROUPS = (YW / 16) / 4;     // 5 groups of 4 N-tiles
    int wave = (blockIdx.x * blockDim.x + threadIdx.x) >> 5;
    if (wave >= Mtiles * NGROUPS) return;      // wave-uniform: EXEC all-ones at WMMA
    int lane = threadIdx.x & 31;
    int tm = wave / NGROUPS;
    int n0 = (wave % NGROUPS) * 64;            // first col of the 4-tile sweep
    int m0 = tm * 16;
    int half = lane >> 4, lm = lane & 15;

    // ---- load A fragments once (16-bit A 16x32 layout, packed-pair dwords) ----
    const uint32_t* arow = (const uint32_t*)(Abf + (size_t)(m0 + lm) * K);
    union Frag { v16bf v; uint32_t u[8]; };
    Frag a[KT];
#pragma unroll
    for (int kt = 0; kt < KT; ++kt) {
        int abase = (kt * 16) + (half ? 4 : 0);     // (k0>>1) + half*4
#pragma unroll
        for (int j = 0; j < 4; ++j) a[kt].u[j] = arow[abase + j];
#pragma unroll
        for (int j = 0; j < 4; ++j) a[kt].u[4 + j] = arow[abase + 8 + j];
    }

    // ---- sweep 4 N-tiles with the same A ----
#pragma unroll
    for (int t = 0; t < 4; ++t) {
        int nt0 = n0 + t * 16;
        const uint32_t* brow = (const uint32_t*)(Wt + (size_t)(nt0 + lm) * K);
        v8f c = {};
#pragma unroll
        for (int kt = 0; kt < KT; ++kt) {
            Frag b;
            int bbase = (kt * 16) + (half ? 4 : 0); // (k0>>1) + half*4
#pragma unroll
            for (int j = 0; j < 4; ++j) b.u[j] = brow[bbase + j];
#pragma unroll
            for (int j = 0; j < 4; ++j) b.u[4 + j] = brow[bbase + 8 + j];
            c = __builtin_amdgcn_wmma_f32_16x16x32_bf16(false, a[kt].v, false, b.v,
                                                        (short)0, c, false, false);
        }
        // D f32 16x16: VGPR r -> row r+8*half, col lane%16
        float* yb = Y + (size_t)m0 * YW + nt0 + lm;
#pragma unroll
        for (int r = 0; r < 8; ++r) yb[(size_t)(r + 8 * half) * YW] = c[r];
    }
}

// ---------------- agg init: root path + bias (fully rewrites agg) -----------
__global__ __launch_bounds__(256) void initagg_kernel(const float* __restrict__ Y,
                                                      const float* __restrict__ bias,
                                                      float* __restrict__ agg) {
    int i = blockIdx.x * blockDim.x + threadIdx.x;
    if (i >= NN * COUT) return;
    int node = i >> 5, c = i & 31;
    agg[i] = Y[(size_t)node * YW + 288 + c] + bias[c];
}

// ---------------- edge scatter: one wave per edge, lane = channel -----------
__global__ __launch_bounds__(256) void scatter_kernel(const float* __restrict__ Y,
                                                      const float* __restrict__ Bb,
                                                      const int* __restrict__ ei,
                                                      float* __restrict__ agg) {
    int gid = blockIdx.x * blockDim.x + threadIdx.x;
    int e = gid >> 5;
    if (e >= EE) return;
    int lane = gid & 31;
    int s = ei[e];        // src
    int d = ei[EE + e];   // dst
    const float* yrow = Y + (size_t)s * YW;
    // one basis load per lane (lanes 0..8), broadcast via wave32 shuffle
    float bv = (lane < KK) ? Bb[(size_t)e * KK + lane] : 0.f;
    float acc = 0.f;
#pragma unroll
    for (int k = 0; k < KK; ++k) {
        float bk = __shfl(bv, k, 32);
        acc += bk * yrow[k * 32 + lane];       // 128B coalesced per k, L2-resident
    }
    atomicAdd(agg + (size_t)d * COUT + lane, acc);
}

// ---------------- layer-1 epilogue: relu + concat skip -> bf16 [N,64] -------
__global__ __launch_bounds__(256) void post_concat_kernel(const float* __restrict__ agg,
                                                          const float* __restrict__ skip,
                                                          uint16_t* __restrict__ Hbf64) {
    int i = blockIdx.x * blockDim.x + threadIdx.x;
    if (i >= NN * CIN) return;
    int node = i >> 6, c = i & 63;
    float v = (c < 32) ? fmaxf(agg[(size_t)node * 32 + c], 0.f)
                       : skip[(size_t)node * 32 + (c - 32)];
    Hbf64[i] = f32_to_bf16(v);
}

// ---------------- relu -> bf16 [N,32] ---------------------------------------
__global__ __launch_bounds__(256) void post_relu_bf16_kernel(const float* __restrict__ agg,
                                                             uint16_t* __restrict__ out) {
    int i = blockIdx.x * blockDim.x + threadIdx.x;
    if (i < NN * COUT) out[i] = f32_to_bf16(fmaxf(agg[i], 0.f));
}

// ---------------- relu -> f32 output ---------------------------------------
__global__ __launch_bounds__(256) void post_relu_f32_kernel(const float* __restrict__ agg,
                                                            float* __restrict__ out) {
    int i = blockIdx.x * blockDim.x + threadIdx.x;
    if (i < NN * COUT) out[i] = fmaxf(agg[i], 0.f);
}

extern "C" void kernel_launch(void* const* d_in, const int* in_sizes, int n_in,
                              void* d_out, int out_size, void* d_ws, size_t ws_size,
                              hipStream_t stream) {
    const float* x      = (const float*)d_in[0];
    const int*   ei     = (const int*)d_in[1];     // [2,E] flat: src=ei[e], dst=ei[E+e]
    const float* pseudo = (const float*)d_in[2];
    const float* skip   = (const float*)d_in[3];
    const float* W1     = (const float*)d_in[4];
    const float* root1  = (const float*)d_in[5];
    const float* b1     = (const float*)d_in[6];
    const float* W2     = (const float*)d_in[7];
    const float* root2  = (const float*)d_in[8];
    const float* b2     = (const float*)d_in[9];
    float* out = (float*)d_out;

    // workspace carve-out (all buffers fully rewritten every call)
    char* ws = (char*)d_ws;
    size_t off = 0;
    auto carve = [&](size_t bytes) -> void* {
        void* p = ws + off;
        off += (bytes + 255) & ~(size_t)255;
        return p;
    };
    float*    Bb    = (float*)   carve((size_t)EE * KK * 4);   // 14.4 MB
    uint16_t* Xbf   = (uint16_t*)carve((size_t)NN * CIN * 2);  // 6.4 MB (reused as Hbf64)
    uint16_t* Hbf32 = (uint16_t*)carve((size_t)NN * COUT * 2); // 3.2 MB
    float*    Y     = (float*)   carve((size_t)NN * YW * 4);   // 64 MB (L2-resident)
    float*    agg   = (float*)   carve((size_t)NN * COUT * 4); // 6.4 MB
    uint16_t* Wt1   = (uint16_t*)carve((size_t)YW * CIN * 2);
    uint16_t* Wt2   = (uint16_t*)carve((size_t)YW * COUT * 2);

    const int T = 256;
    const int Mtiles = NN / 16;                      // 3125 exactly
    const int gemmWaves = Mtiles * 5;                // 4 N-tiles per wave, 20 N-tiles total
    const int gemmBlocks = (gemmWaves * 32 + T - 1) / T;
    const int scatBlocks = (EE * 32) / T;            // 50000 exactly
    const int aggBlocks  = (NN * COUT + T - 1) / T;

    // one-time per call prep
    basis_kernel<<<(EE + T - 1) / T, T, 0, stream>>>(pseudo, Bb);
    cvt_bf16_kernel<<<(NN * CIN + T - 1) / T, T, 0, stream>>>(x, Xbf, NN * CIN);
    build_wt_kernel<<<(YW * CIN + T - 1) / T, T, 0, stream>>>(W1, root1, Wt1, CIN);
    build_wt_kernel<<<(YW * COUT + T - 1) / T, T, 0, stream>>>(W2, root2, Wt2, COUT);

    // ---- layer 1: x[N,64] -> agg[N,32] ----
    gemm_wmma_kernel<CIN><<<gemmBlocks, T, 0, stream>>>(Xbf, Wt1, Y, Mtiles);
    initagg_kernel<<<aggBlocks, T, 0, stream>>>(Y, b1, agg);
    scatter_kernel<<<scatBlocks, T, 0, stream>>>(Y, Bb, ei, agg);
    post_concat_kernel<<<(NN * CIN + T - 1) / T, T, 0, stream>>>(agg, skip, Xbf); // -> h[N,64] bf16

    // ---- layer 2: h[N,64] -> agg[N,32] (same W1/root1/b1) ----
    gemm_wmma_kernel<CIN><<<gemmBlocks, T, 0, stream>>>(Xbf, Wt1, Y, Mtiles);
    initagg_kernel<<<aggBlocks, T, 0, stream>>>(Y, b1, agg);
    scatter_kernel<<<scatBlocks, T, 0, stream>>>(Y, Bb, ei, agg);
    post_relu_bf16_kernel<<<aggBlocks, T, 0, stream>>>(agg, Hbf32);               // -> [N,32] bf16

    // ---- layer 3: [N,32] -> out[N,32] (W2/root2/b2) ----
    gemm_wmma_kernel<COUT><<<gemmBlocks, T, 0, stream>>>(Hbf32, Wt2, Y, Mtiles);
    initagg_kernel<<<aggBlocks, T, 0, stream>>>(Y, b2, agg);
    scatter_kernel<<<scatBlocks, T, 0, stream>>>(Y, Bb, ei, agg);
    post_relu_f32_kernel<<<aggBlocks, T, 0, stream>>>(agg, out);
}
